// SingleHeadAttention_72662256714106
// MI455X (gfx1250) — matrix operand order repaired and checked by reading, not compiled, not run on hardware
//
#include <hip/hip_runtime.h>

typedef __attribute__((ext_vector_type(16))) _Float16 v16h;
typedef __attribute__((ext_vector_type(8)))  _Float16 v8h;
typedef __attribute__((ext_vector_type(8)))  float    v8f;

#define D_MODEL 512
#define D_K     64
#define SEQ     4096
#define BATCH   4
#define NROWS   (BATCH * SEQ)
#define TILES   (NROWS / 16)          // 1024 query tiles of 16 rows
#define WAVES_PER_BLOCK 8             // 256 threads

union FragH { v16h v; v8h h[2]; };

__device__ inline v8f zero8() {
  v8f z = {0.f, 0.f, 0.f, 0.f, 0.f, 0.f, 0.f, 0.f};
  return z;
}

// A-operand (16x32 f16): lane m = l&15, kb = 8*(l>>4); K = {kb..kb+7, kb+16..kb+23}
__device__ inline v16h load_fragA_h(const _Float16* base, int ld) {
  const int lane = threadIdx.x & 31;
  const int m  = lane & 15;
  const int kb = (lane >> 4) << 3;
  const _Float16* p = base + (size_t)m * ld + kb;
  FragH f;
  f.h[0] = *(const v8h*)(p);
  f.h[1] = *(const v8h*)(p + 16);
  return f.v;
}

__device__ inline v16h load_fragA_f32(const float* base, int ld) {
  const int lane = threadIdx.x & 31;
  const int m  = lane & 15;
  const int kb = (lane >> 4) << 3;
  const float* p = base + (size_t)m * ld + kb;
  v16h r;
#pragma unroll
  for (int j = 0; j < 8; ++j) {
    r[j]     = (_Float16)p[j];
    r[j + 8] = (_Float16)p[16 + j];
  }
  return r;
}

// B-operand (32x16 f16): lane n = l&15, kb = 8*(l>>4); K = {kb..kb+7, kb+16..kb+23}.
// "base" points at the transposed-B matrix stored row-major with rows = N, cols = K.
__device__ inline v16h load_fragB_h(const _Float16* base, int ld) {
  const int lane = threadIdx.x & 31;
  const int n  = lane & 15;
  const int kb = (lane >> 4) << 3;
  const _Float16* p = base + (size_t)n * ld + kb;
  FragH f;
  f.h[0] = *(const v8h*)(p);
  f.h[1] = *(const v8h*)(p + 16);
  return f.v;
}

#define WMMA_F16(A, B, C) \
  __builtin_amdgcn_wmma_f32_16x16x32_f16(false, (A), false, (B), (short)0, (C), false, false)

// ---------------------------------------------------------------------------
// Kernel 0: one-shot f32 -> f16 conversion of the three weight matrices
// (64x512 each). Removes all per-tile weight conversion VALU from kernel 1.
// ---------------------------------------------------------------------------
__global__ __launch_bounds__(256) void wcvt_kernel(
    const float* __restrict__ Wq, const float* __restrict__ Wk,
    const float* __restrict__ Wv,
    _Float16* __restrict__ Whq, _Float16* __restrict__ Whk,
    _Float16* __restrict__ Whv) {
  const int i = blockIdx.x * 256 + threadIdx.x;   // 0 .. 32767
  Whq[i] = (_Float16)Wq[i];
  Whk[i] = (_Float16)Wk[i];
  Whv[i] = (_Float16)Wv[i];
}

// ---------------------------------------------------------------------------
// Kernel 1: Q = (x @ Wq^T) * 1/sqrt(d_k) -> f16 ; K = x @ Wk^T -> f16 ;
//           V^T (per batch, d_k x S)     -> f16
// One wave handles one 16-row tile of all three projections.
// ---------------------------------------------------------------------------
__global__ __launch_bounds__(256) void qkv_proj_kernel(
    const float* __restrict__ x,
    const _Float16* __restrict__ Whq, const _Float16* __restrict__ Whk,
    const _Float16* __restrict__ Whv,
    _Float16* __restrict__ Qh, _Float16* __restrict__ Kh,
    _Float16* __restrict__ Vth) {
  const int wave = threadIdx.x >> 5;
  const int lane = threadIdx.x & 31;
  const int N    = lane & 15;
  const int half = lane >> 4;
  const int tile = blockIdx.x * WAVES_PER_BLOCK + wave;
  const int r0   = tile * 16;

  v8f cQ[4], cK[4], cV[4];
#pragma unroll
  for (int nt = 0; nt < 4; ++nt) { cQ[nt] = zero8(); cK[nt] = zero8(); cV[nt] = zero8(); }

  for (int kc = 0; kc < D_MODEL; kc += 32) {
    v16h a = load_fragA_f32(x + (size_t)r0 * D_MODEL + kc, D_MODEL);
#pragma unroll
    for (int nt = 0; nt < 4; ++nt) {
      const size_t woff = (size_t)(nt * 16) * D_MODEL + kc;
      cQ[nt] = WMMA_F16(a, load_fragB_h(Whq + woff, D_MODEL), cQ[nt]);
      cK[nt] = WMMA_F16(a, load_fragB_h(Whk + woff, D_MODEL), cK[nt]);
      cV[nt] = WMMA_F16(a, load_fragB_h(Whv + woff, D_MODEL), cV[nt]);
    }
  }

  const int b = r0 / SEQ;
  const int s = r0 % SEQ;
  const float qscale = 0.125f;  // 1/sqrt(64), folded into Q
#pragma unroll
  for (int nt = 0; nt < 4; ++nt) {
#pragma unroll
    for (int r = 0; r < 8; ++r) {
      const int M = r + 8 * half;
      const size_t row = (size_t)(r0 + M);
      Qh[row * D_K + nt * 16 + N] = (_Float16)(cQ[nt][r] * qscale);
      Kh[row * D_K + nt * 16 + N] = (_Float16)(cK[nt][r]);
      Vth[((size_t)b * D_K + nt * 16 + N) * SEQ + (size_t)(s + M)] = (_Float16)(cV[nt][r]);
    }
  }
}

// ---------------------------------------------------------------------------
// Kernel 2: causal flash attention. One wave owns a 16-query tile; streams
// 64-key chunks: S = Q K^T (up to 8 WMMAs), online softmax in f32, P staged
// through LDS to A-fragment layout, O += P V (up to 8 WMMAs). Tiles are
// processed heaviest-first (reverse order) for load balance under causality.
// ---------------------------------------------------------------------------
__global__ __launch_bounds__(256) void attn_kernel(
    const _Float16* __restrict__ Qh, const _Float16* __restrict__ Kh,
    const _Float16* __restrict__ Vth, float* __restrict__ out) {
  __shared__ __align__(16) _Float16 pbuf[WAVES_PER_BLOCK][16 * 72];  // 16x64 tile, padded ld=72

  const int wave = threadIdx.x >> 5;
  const int lane = threadIdx.x & 31;
  const int N    = lane & 15;
  const int half = lane >> 4;
  const int tile = (TILES - 1) - (blockIdx.x * WAVES_PER_BLOCK + wave);  // heavy first
  const int b    = (tile * 16) / SEQ;
  const int q0   = (tile * 16) % SEQ;

  const _Float16* Qb = Qh + (size_t)(b * SEQ + q0) * D_K;
  const v16h qa0 = load_fragA_h(Qb, D_K);        // k-dim 0..31
  const v16h qa1 = load_fragA_h(Qb + 32, D_K);   // k-dim 32..63

  v8f o[4];
#pragma unroll
  for (int nt = 0; nt < 4; ++nt) o[nt] = zero8();
  float mi[8], li[8];
#pragma unroll
  for (int r = 0; r < 8; ++r) { mi[r] = -3.0e38f; li[r] = 0.f; }

  _Float16* pl = &pbuf[wave][0];

  for (int s0 = 0; s0 < q0 + 16; s0 += 64) {
    // ---- scores: 16 q x 64 keys, accumulated over d_k = 64 ----
    v8f sc[4];
#pragma unroll
    for (int nt = 0; nt < 4; ++nt) sc[nt] = zero8();
#pragma unroll
    for (int nt = 0; nt < 4; ++nt) {
      if (s0 + nt * 16 < q0 + 16) {          // wave-uniform: skip fully-masked groups
        const _Float16* Kb = Kh + (size_t)(b * SEQ + s0 + nt * 16) * D_K;
        sc[nt] = WMMA_F16(qa0, load_fragB_h(Kb, D_K), sc[nt]);
        sc[nt] = WMMA_F16(qa1, load_fragB_h(Kb + 32, D_K), sc[nt]);
      }
    }

    // ---- prefetch next chunk's K rows and V^T segments (global_prefetch_b8) ----
    const int s0n = s0 + 64;
    if (s0n < q0 + 16 && s0n + 64 <= SEQ) {
      __builtin_prefetch(Kh + (size_t)(b * SEQ + s0n + lane * 2) * D_K, 0, 1);
      __builtin_prefetch(Vth + ((size_t)b * D_K + lane * 2) * SEQ + s0n, 0, 1);
    }

    // ---- online softmax; P written to LDS as it is produced ----
#pragma unroll
    for (int r = 0; r < 8; ++r) {
      const int M = r + 8 * half;
      const int q = q0 + M;
      float sv[4];
#pragma unroll
      for (int nt = 0; nt < 4; ++nt)
        sv[nt] = ((s0 + nt * 16 + N) <= q) ? sc[nt][r] : -3.0e38f;  // causal mask
      float rm = fmaxf(fmaxf(sv[0], sv[1]), fmaxf(sv[2], sv[3]));
#pragma unroll
      for (int off = 1; off < 16; off <<= 1)
        rm = fmaxf(rm, __shfl_xor(rm, off, 32));     // reduce across 16-lane half
      const float newm = fmaxf(mi[r], rm);
      float ps = 0.f;
#pragma unroll
      for (int nt = 0; nt < 4; ++nt) {
        const float p = __expf(sv[nt] - newm);
        ps += p;
        pl[M * 72 + nt * 16 + N] = (_Float16)p;
      }
#pragma unroll
      for (int off = 1; off < 16; off <<= 1)
        ps += __shfl_xor(ps, off, 32);
      const float rescale = __expf(mi[r] - newm);
      li[r] = li[r] * rescale + ps;
      mi[r] = newm;
#pragma unroll
      for (int nt = 0; nt < 4; ++nt) o[nt][r] *= rescale;
    }
    asm volatile("s_wait_dscnt 0" ::: "memory");     // wave-local LDS RAW fence

    // ---- O += P @ V (B-operand rows contiguous in V^T) ----
    const v16h pa0 = load_fragA_h(pl, 72);           // keys 0..31
    const bool hiHalf = (s0 + 32) < q0 + 16;         // keys 32..63 carry nonzero P?
    v16h pa1;
    if (hiHalf) pa1 = load_fragA_h(pl + 32, 72);     // keys 32..63
#pragma unroll
    for (int nt = 0; nt < 4; ++nt) {
      const _Float16* Vb = Vth + ((size_t)b * D_K + nt * 16) * SEQ + s0;
      o[nt] = WMMA_F16(pa0, load_fragB_h(Vb, SEQ), o[nt]);
      if (hiHalf)
        o[nt] = WMMA_F16(pa1, load_fragB_h(Vb + 32, SEQ), o[nt]);
    }
    asm volatile("s_wait_dscnt 0" ::: "memory");     // WAR guard before next chunk's stores
  }

  // ---- normalize and store (f32 output) ----
#pragma unroll
  for (int nt = 0; nt < 4; ++nt) {
#pragma unroll
    for (int r = 0; r < 8; ++r) {
      const int M = r + 8 * half;
      out[(size_t)(b * SEQ + q0 + M) * D_K + nt * 16 + N] = o[nt][r] / li[r];
    }
  }
}

extern "C" void kernel_launch(void* const* d_in, const int* in_sizes, int n_in,
                              void* d_out, int out_size, void* d_ws, size_t ws_size,
                              hipStream_t stream) {
  (void)in_sizes; (void)n_in; (void)out_size; (void)ws_size;
  const float* x  = (const float*)d_in[0];
  const float* Wq = (const float*)d_in[1];
  const float* Wk = (const float*)d_in[2];
  const float* Wv = (const float*)d_in[3];
  float* out = (float*)d_out;

  // Workspace: Q f16 (2 MB) | K f16 (2 MB) | V^T f16 (2 MB) | Wq/Wk/Wv f16 (64 KB each)
  _Float16* Qh  = (_Float16*)d_ws;
  _Float16* Kh  = Qh + (size_t)NROWS * D_K;
  _Float16* Vth = Kh + (size_t)NROWS * D_K;
  _Float16* Whq = Vth + (size_t)NROWS * D_K;
  _Float16* Whk = Whq + (size_t)D_K * D_MODEL;
  _Float16* Whv = Whk + (size_t)D_K * D_MODEL;

  wcvt_kernel<<<(D_K * D_MODEL) / 256, 256, 0, stream>>>(Wq, Wk, Wv, Whq, Whk, Whv);

  const int blocks = TILES / WAVES_PER_BLOCK;  // 128 blocks x 256 threads
  qkv_proj_kernel<<<blocks, 256, 0, stream>>>(x, Whq, Whk, Whv, Qh, Kh, Vth);
  attn_kernel<<<blocks, 256, 0, stream>>>(Qh, Kh, Vth, out);
}
